// Loss_73315091742882
// MI455X (gfx1250) — matrix operand order repaired and checked by reading, not compiled, not run on hardware
//
#include <hip/hip_runtime.h>
#include <hip/hip_bf16.h>
#include <stddef.h>

// ---------------------------------------------------------------------------
// Problem constants (match reference)
// ---------------------------------------------------------------------------
#define B_DIM   4096
#define C_DIM   1000
#define D_DIM   256
#define N2      8192          // 2*B rows in InfoNCE
#define INV_T   (1.0f / 0.07f)
#define CMAX    (1.0f / 0.07f) // logits <= 1/temp since features are unit-norm

#define NTILES          (N2 / 16)            // 512 column tiles
#define CHUNKS          8                    // column split for occupancy
#define TILES_PER_CHUNK (NTILES / CHUNKS)    // 64
#define TILE_BYTES      (D_DIM * 16 * 2)     // 8 KB per column tile (K x 16 bf16)

typedef __attribute__((ext_vector_type(16))) __bf16 v16bf;
typedef __attribute__((ext_vector_type(8)))  __bf16 v8bf;
typedef __attribute__((ext_vector_type(8)))  float  v8f;

// ---------------------------------------------------------------------------
// Block reduction helpers (blockDim.x == 256)
// ---------------------------------------------------------------------------
__device__ __forceinline__ float blockReduceSum(float v, float* red) {
    int tid = threadIdx.x;
    __syncthreads();
    red[tid] = v;
    __syncthreads();
    for (int s = 128; s > 0; s >>= 1) {
        if (tid < s) red[tid] += red[tid + s];
        __syncthreads();
    }
    float r = red[0];
    __syncthreads();
    return r;
}

__device__ __forceinline__ float blockReduceMax(float v, float* red) {
    int tid = threadIdx.x;
    __syncthreads();
    red[tid] = v;
    __syncthreads();
    for (int s = 128; s > 0; s >>= 1) {
        if (tid < s) red[tid] = fmaxf(red[tid], red[tid + s]);
        __syncthreads();
    }
    float r = red[0];
    __syncthreads();
    return r;
}

// ---------------------------------------------------------------------------
// Kernel 0: zero accumulators + per-row InfoNCE partial-sum array
// ---------------------------------------------------------------------------
__global__ void init_ws_kernel(float* acc, float* rowsumArr) {
    int i = blockIdx.x * blockDim.x + threadIdx.x;
    if (i < N2) rowsumArr[i] = 0.0f;
    if (i < 8)  acc[i] = 0.0f;
}

// ---------------------------------------------------------------------------
// Kernel 1: per-row KL + CE + focal on `output` / `master_net_pred`
// ---------------------------------------------------------------------------
__global__ __launch_bounds__(256) void rowloss_kernel(
    const float* __restrict__ out, const float* __restrict__ mast,
    const int* __restrict__ tgt, float* acc) {
    __shared__ float so[C_DIM];
    __shared__ float sm[C_DIM];
    __shared__ float red[256];

    const int r = blockIdx.x;
    const float* orow = out  + (size_t)r * C_DIM;
    const float* mrow = mast + (size_t)r * C_DIM;

    float lmo = -1e30f, lmm = -1e30f;
    for (int i = threadIdx.x; i < C_DIM; i += 256) {
        float o = orow[i], m = mrow[i];
        so[i] = o; sm[i] = m;
        lmo = fmaxf(lmo, o);
        lmm = fmaxf(lmm, m);
    }
    float Mq = blockReduceMax(lmo, red);
    float Mp = blockReduceMax(lmm, red);

    float s1 = 0.f, sTq = 0.f, sTp = 0.f, scr = 0.f;
    for (int i = threadIdx.x; i < C_DIM; i += 256) {
        float o = so[i], m = sm[i];
        s1  += __expf(o - Mq);
        sTq += __expf((o - Mq) * 0.25f);
        float ep = __expf((m - Mp) * 0.25f);
        sTp += ep;
        scr += ep * (m - o) * 0.25f;
    }
    s1  = blockReduceSum(s1,  red);
    sTq = blockReduceSum(sTq, red);
    sTp = blockReduceSum(sTp, red);
    scr = blockReduceSum(scr, red);

    if (threadIdx.x == 0) {
        float lse1  = Mq + __logf(s1);
        float klrow = scr / sTp
                    + (Mq * 0.25f + __logf(sTq))
                    - (Mp * 0.25f + __logf(sTp));
        float ot    = so[tgt[r]];
        float logpt = ot - lse1;
        float pt    = __expf(logpt);
        float gamma = (pt < 0.2f) ? 5.0f : ((pt < 0.5f) ? 3.0f : 1.0f);
        float foc   = -__powf(fmaxf(1.0f - pt, 0.0f), gamma) * logpt;
        atomicAdd(&acc[0], klrow);
        atomicAdd(&acc[1], -logpt);   // CE
        atomicAdd(&acc[2], foc);
    }
}

// ---------------------------------------------------------------------------
// Kernel 2: adversarial cross-entropy rows
// ---------------------------------------------------------------------------
__global__ __launch_bounds__(256) void advce_kernel(
    const float* __restrict__ out, const int* __restrict__ tgt, float* acc) {
    __shared__ float so[C_DIM];
    __shared__ float red[256];

    const int r = blockIdx.x;
    const float* orow = out + (size_t)r * C_DIM;

    float lmo = -1e30f;
    for (int i = threadIdx.x; i < C_DIM; i += 256) {
        float o = orow[i];
        so[i] = o;
        lmo = fmaxf(lmo, o);
    }
    float Mq = blockReduceMax(lmo, red);

    float s1 = 0.f;
    for (int i = threadIdx.x; i < C_DIM; i += 256)
        s1 += __expf(so[i] - Mq);
    s1 = blockReduceSum(s1, red);

    if (threadIdx.x == 0) {
        float lse1 = Mq + __logf(s1);
        atomicAdd(&acc[3], lse1 - so[tgt[r]]);
    }
}

// ---------------------------------------------------------------------------
// Kernel 3: L2-normalize concat(feat, feat_masked) -> bf16 Fhat [8192,256]
// plus a TILED copy FhatT_t[col_tile][16][K=256] so that a 16-column
// WMMA B tile is one contiguous 8 KB block (fully coalesced LDS fill).
// ---------------------------------------------------------------------------
__global__ __launch_bounds__(256) void normalize_kernel(
    const float* __restrict__ fp, const float* __restrict__ fpm,
    __bf16* __restrict__ Fhat, __bf16* __restrict__ FhatT) {
    __shared__ float red[256];
    const int r = blockIdx.x;                       // 0..8191
    const float* src = (r < B_DIM) ? (fp  + (size_t)r * D_DIM)
                                   : (fpm + (size_t)(r - B_DIM) * D_DIM);
    float x  = src[threadIdx.x];
    float ss = blockReduceSum(x * x, red);
    float rn = rsqrtf(ss);
    __bf16 v = (__bf16)(x * rn);
    Fhat[(size_t)r * D_DIM + threadIdx.x] = v;
    // tiled layout: tile = r/16, row = col-in-tile (=r%16), k (=threadIdx.x) contiguous
    FhatT[(size_t)(r >> 4) * (D_DIM * 16)
          + (size_t)(r & 15) * D_DIM + threadIdx.x] = v;
}

// ---------------------------------------------------------------------------
// Kernel 4: fused Gram + exp-sum (flash-style InfoNCE), column-chunked.
// grid = (64 row-groups of 128 rows, 8 column chunks); 256 thr = 8 wave32.
// - A fragments (16 rows x 256 bf16) resident in registers per wave.
// - B tiles double-buffered through LDS, filled cooperatively by the block
//   (one coalesced 32-byte load per thread) and shared by all 8 waves -> 8x less L2.
// - 8 chained v_wmma_f32_16x16x32_bf16 per tile split over 2 accumulators.
// - Fixed softmax max = 1/0.07 (unit-norm features) => column splits are
//   exact partial sums; diagonal masked exactly; positive logit captured
//   in-flight.
// ---------------------------------------------------------------------------
__global__ __launch_bounds__(256) void nce_gram_kernel(
    const __bf16* __restrict__ Fhat, const __bf16* __restrict__ FhatT,
    float* __restrict__ rowsumArr, float* __restrict__ posArr) {
    __shared__ __align__(32) __bf16 smem[2][D_DIM * 16];   // 2 x 8 KB

    const int tid  = threadIdx.x;
    const int lane = tid & 31;
    const int wv   = tid >> 5;                     // wave in block: 0..7
    const int r0   = blockIdx.x * 128 + wv * 16;   // this wave's row stripe
    const int ct0  = blockIdx.y * TILES_PER_CHUNK; // first column tile
    const int tx   = lane & 15;                    // N index / A row
    const int half = lane >> 4;                    // K-half (A) / M-offset (C)

    // ---- A fragments for all 8 k-steps (ISA 16-bit A layout) ----
    v16bf a[8];
    {
        const __bf16* arow = Fhat + (size_t)(r0 + tx) * D_DIM;
        #pragma unroll
        for (int ks = 0; ks < 8; ++ks) {
            v8bf lo = *(const v8bf*)(arow + ks * 32 + half * 8);
            v8bf hi = *(const v8bf*)(arow + ks * 32 + 16 + half * 8);
            #pragma unroll
            for (int j = 0; j < 8; ++j) { a[ks][j] = lo[j]; a[ks][j + 8] = hi[j]; }
        }
    }

    float rowsum[8];
    float posv[8];
    #pragma unroll
    for (int v = 0; v < 8; ++v) { rowsum[v] = 0.0f; posv[v] = -1e30f; }

    // ---- preload first tile into LDS buffer 0 ----
    v16bf stage = *(const v16bf*)(FhatT + (size_t)ct0 * (D_DIM * 16) + tid * 16);
    *(v16bf*)&smem[0][tid * 16] = stage;
    __syncthreads();

    for (int it = 0; it < TILES_PER_CHUNK; ++it) {
        const int ct  = ct0 + it;
        const int cur = it & 1;

        // issue next tile's global load early (hidden behind WMMAs)
        if (it + 1 < TILES_PER_CHUNK)
            stage = *(const v16bf*)(FhatT + (size_t)(ct + 1) * (D_DIM * 16) + tid * 16);
        if (it + 2 < TILES_PER_CHUNK)
            __builtin_prefetch((const void*)(FhatT + (size_t)(ct + 2) * (D_DIM * 16) + tid * 16), 0, 3);

        // bulk-read B fragments from LDS (clause of ds_load_b128s, one wait)
        v16bf b[8];
        #pragma unroll
        for (int ks = 0; ks < 8; ++ks)
        {
            v8bf blo = *(const v8bf*)&smem[cur][tx * D_DIM + ks * 32 + half * 8];
            v8bf bhi = *(const v8bf*)&smem[cur][tx * D_DIM + ks * 32 + 16 + half * 8];
            #pragma unroll
            for (int j = 0; j < 8; ++j) { b[ks][j] = blo[j]; b[ks][j + 8] = bhi[j]; }
        }

        // 8 WMMAs over K=256, two independent accumulator chains
        v8f c0 = {}, c1 = {};
        c0 = __builtin_amdgcn_wmma_f32_16x16x32_bf16(false, a[0], false, b[0], (short)0, c0, false, false);
        c1 = __builtin_amdgcn_wmma_f32_16x16x32_bf16(false, a[1], false, b[1], (short)0, c1, false, false);
        c0 = __builtin_amdgcn_wmma_f32_16x16x32_bf16(false, a[2], false, b[2], (short)0, c0, false, false);
        c1 = __builtin_amdgcn_wmma_f32_16x16x32_bf16(false, a[3], false, b[3], (short)0, c1, false, false);
        c0 = __builtin_amdgcn_wmma_f32_16x16x32_bf16(false, a[4], false, b[4], (short)0, c0, false, false);
        c1 = __builtin_amdgcn_wmma_f32_16x16x32_bf16(false, a[5], false, b[5], (short)0, c1, false, false);
        c0 = __builtin_amdgcn_wmma_f32_16x16x32_bf16(false, a[6], false, b[6], (short)0, c0, false, false);
        c1 = __builtin_amdgcn_wmma_f32_16x16x32_bf16(false, a[7], false, b[7], (short)0, c1, false, false);

        // fold the 16x16 tile into per-row exp-sums
        const int col = ct * 16 + tx;
        #pragma unroll
        for (int v = 0; v < 8; ++v) {
            const int row = r0 + v + half * 8;       // C layout: M = v + 8*half
            float logit   = (c0[v] + c1[v]) * INV_T;
            float e       = (col == row) ? 0.0f : __expf(logit - CMAX);
            rowsum[v] += e;
            if (col == ((row + B_DIM) & (N2 - 1))) posv[v] = logit;
        }

        // stash next tile into the other LDS buffer
        if (it + 1 < TILES_PER_CHUNK)
            *(v16bf*)&smem[cur ^ 1][tid * 16] = stage;
        __syncthreads();
    }

    // reduce each row across its 16 lanes (same half, tx varies)
    #pragma unroll
    for (int v = 0; v < 8; ++v) {
        float s = rowsum[v], p = posv[v];
        #pragma unroll
        for (int m = 8; m >= 1; m >>= 1) {
            s += __shfl_xor(s, m, 16);
            p  = fmaxf(p, __shfl_xor(p, m, 16));
        }
        rowsum[v] = s; posv[v] = p;
    }
    if (tx == 0) {  // lanes 0 and 16: rows r0..r0+7 and r0+8..r0+15
        #pragma unroll
        for (int v = 0; v < 8; ++v) {
            const int row = r0 + v + half * 8;
            atomicAdd(&rowsumArr[row], rowsum[v]);
            const int partner = (row + B_DIM) & (N2 - 1);
            const int pct = partner >> 4;
            if (pct >= ct0 && pct < ct0 + TILES_PER_CHUNK)
                posArr[row] = posv[v];               // exactly one chunk hits
        }
    }
}

// ---------------------------------------------------------------------------
// Kernel 5: per-row lse - pos, reduced into acc[4]
// ---------------------------------------------------------------------------
__global__ __launch_bounds__(256) void nce_reduce_kernel(
    const float* __restrict__ rowsumArr, const float* __restrict__ posArr,
    float* acc) {
    __shared__ float red[256];
    const int i = blockIdx.x * 256 + threadIdx.x;
    float term = (CMAX + __logf(rowsumArr[i])) - posArr[i];
    float s = blockReduceSum(term, red);
    if (threadIdx.x == 0) atomicAdd(&acc[4], s);
}

// ---------------------------------------------------------------------------
// Kernel 6: combine with reference weights into d_out[0]
//   loss = 1*(KL*0.5*16/(B*C) + CE*0.5/B) + NCE/2B + FOC/B + ADV/B
// ---------------------------------------------------------------------------
__global__ void finalize_kernel(const float* acc, float* out) {
    if (threadIdx.x == 0 && blockIdx.x == 0) {
        float kl  = acc[0], ce = acc[1], foc = acc[2], adv = acc[3], nce = acc[4];
        float loss_main = kl * (0.5f * 16.0f) / ((float)B_DIM * (float)C_DIM)
                        + ce * 0.5f / (float)B_DIM;
        float loss = loss_main
                   + nce / (float)N2
                   + foc / (float)B_DIM
                   + adv / (float)B_DIM;
        out[0] = loss;
    }
}

// ---------------------------------------------------------------------------
// Launch
// ---------------------------------------------------------------------------
extern "C" void kernel_launch(void* const* d_in, const int* in_sizes, int n_in,
                              void* d_out, int out_size, void* d_ws, size_t ws_size,
                              hipStream_t stream) {
    (void)in_sizes; (void)n_in; (void)out_size; (void)ws_size;
    const float* output      = (const float*)d_in[0];
    const int*   target      = (const int*)  d_in[1];
    const float* master_pred = (const float*)d_in[2];
    const float* feat        = (const float*)d_in[3];
    const float* feat_masked = (const float*)d_in[4];
    const float* output_adv  = (const float*)d_in[5];
    const int*   target_adv  = (const int*)  d_in[6];

    char* ws = (char*)d_ws;
    const size_t fhat_bytes = (size_t)N2 * D_DIM * 2;       // 4 MB
    __bf16* Fhat      = (__bf16*)(ws);
    __bf16* FhatT     = (__bf16*)(ws + fhat_bytes);
    float*  rowsumArr = (float*) (ws + 2 * fhat_bytes);                  // 32 KB
    float*  posArr    = (float*) (ws + 2 * fhat_bytes + N2 * 4);         // 32 KB
    float*  acc       = (float*) (ws + 2 * fhat_bytes + 2 * N2 * 4);     // 32 B

    init_ws_kernel <<<N2 / 256, 256, 0, stream>>>(acc, rowsumArr);
    rowloss_kernel <<<B_DIM, 256, 0, stream>>>(output, master_pred, target, acc);
    advce_kernel   <<<B_DIM, 256, 0, stream>>>(output_adv, target_adv, acc);
    normalize_kernel<<<N2, 256, 0, stream>>>(feat, feat_masked, Fhat, FhatT);
    nce_gram_kernel <<<dim3(N2 / 128, CHUNKS), 256, 0, stream>>>(Fhat, FhatT,
                                                                 rowsumArr, posArr);
    nce_reduce_kernel<<<N2 / 256, 256, 0, stream>>>(rowsumArr, posArr, acc);
    finalize_kernel <<<1, 32, 0, stream>>>(acc, (float*)d_out);
}
